// Uncertainty_50663434224253
// MI455X (gfx1250) — matrix operand order repaired and checked by reading, not compiled, not run on hardware
//
#include <hip/hip_runtime.h>
#include <math.h>

typedef __attribute__((ext_vector_type(16))) _Float16 v16h;
typedef __attribute__((ext_vector_type(8)))  _Float16 h8;
typedef __attribute__((ext_vector_type(8)))  float    v8f;

#define IMG_H 128
#define IMG_W 128
#define BATCH 64
#define HW (IMG_H*IMG_W)
#define EPS 1e-5f
#define ELU_ADD 1.0000001f

// tap -> LDS position offset (row*136 + col), clamped; constant after unroll
#define TAP3(t) ((((t) > 80 ? 80 : (t)) / 9) * 136 + (((t) > 80 ? 80 : (t)) % 9))
#define TAP2(t) ((((t) > 8  ? 8  : (t)) / 3) * 136 + (((t) > 8  ? 8  : (t)) % 3))

#define SHUF16(a,b) __builtin_shufflevector(a, b, 0,1,2,3,4,5,6,7,8,9,10,11,12,13,14,15)

// ---------------------------------------------------------------------------
// zero the stat accumulators each launch (deterministic graph replay)
// ---------------------------------------------------------------------------
__global__ void init_acc_k(float* acc) {
    int i = threadIdx.x;
    if (i < 112) acc[i] = 0.f;
}

// ---------------------------------------------------------------------------
// conv1: 4->8, k=9, pad=4, groups=2. Direct f32 (only 2 in-ch/group; BW bound).
// ---------------------------------------------------------------------------
__global__ __launch_bounds__(256) void conv1_k(
    const float* __restrict__ x, const float* __restrict__ w1,
    const float* __restrict__ b1, float* __restrict__ h1, float* __restrict__ acc1)
{
    int plane = blockIdx.x;
    int o = plane & 7;
    int b = plane >> 3;
    int pix = blockIdx.y * 256 + threadIdx.x;
    int hh = pix >> 7, ww = pix & 127;
    int gr = o >> 2;
    float s = b1[o];
    #pragma unroll
    for (int ci = 0; ci < 2; ++ci) {
        int cin = gr * 2 + ci;
        const float* xp = x + ((size_t)b * 4 + cin) * HW;
        const float* wp = w1 + (o * 2 + ci) * 81;
        for (int kh = 0; kh < 9; ++kh) {
            int hin = hh + kh - 4;
            if (hin < 0 || hin >= IMG_H) continue;
            for (int kw = 0; kw < 9; ++kw) {
                int win = ww + kw - 4;
                if (win < 0 || win >= IMG_W) continue;
                s += xp[hin * IMG_W + win] * wp[kh * 9 + kw];
            }
        }
    }
    h1[((size_t)b * 8 + o) * HW + pix] = s;

    __shared__ float rs[2];
    if (threadIdx.x == 0) { rs[0] = 0.f; rs[1] = 0.f; }
    __syncthreads();
    atomicAdd(&rs[0], s);
    atomicAdd(&rs[1], s * s);
    __syncthreads();
    if (threadIdx.x == 0) {
        atomicAdd(&acc1[o], rs[0]);
        atomicAdd(&acc1[8 + o], rs[1]);
    }
}

// ---------------------------------------------------------------------------
// finalize BN stats: scale = gamma*rsqrt(var+eps), shift = beta - mean*scale
// ---------------------------------------------------------------------------
__global__ void finalize_k(const float* __restrict__ acc,
                           const float* __restrict__ gamma,
                           const float* __restrict__ beta,
                           float* __restrict__ sc, float* __restrict__ sh, int C)
{
    int c = threadIdx.x;
    if (c >= C) return;
    const float N = (float)((size_t)BATCH * HW);
    float mean = acc[c] / N;
    float var  = acc[C + c] / N - mean * mean;
    float rstd = rsqrtf(var + EPS);
    float s = gamma[c] * rstd;
    sc[c] = s;
    sh[c] = beta[c] - mean * s;
}

// ---------------------------------------------------------------------------
// conv2: 8->16, k=3, pad=1 — WMMA f16 implicit GEMM, channel-interleaved LDS.
// One block = (b, row-pair): stages 4 rows, computes 2 output rows.
// Wave w: row h0+(w>>2), 32-col strip (w&3)*32, two interleaved acc chains.
// ---------------------------------------------------------------------------
__global__ __launch_bounds__(256) void conv2_wmma_k(
    const float* __restrict__ h1, const float* __restrict__ w2,
    const float* __restrict__ b2, const float* __restrict__ sc1,
    const float* __restrict__ sh1, const float* __restrict__ a1,
    float* __restrict__ h2, float* __restrict__ acc2)
{
    __shared__ h8       sA8[4 * 136];      // [row][col] -> 8 channels (f16)
    __shared__ h8       sW8[16 * 12];      // [n][tap(pad 12)] -> 8 channels
    __shared__ float    sstat[32];

    int blk = blockIdx.x;                  // b*64 + hp
    int hp = blk & 63;
    int b  = blk >> 6;
    int h0 = hp * 2;
    int tid = threadIdx.x;
    float alpha = a1[0];

    if (tid < 32) sstat[tid] = 0.f;

    _Float16* sAh = (_Float16*)sA8;
    _Float16* sWh = (_Float16*)sW8;

    // stage activations: 4 rows (h0-1 .. h0+2) x 136 cols x 8 ch, BN1+PReLU
    for (int idx = tid; idx < 4 * 136 * 8; idx += 256) {
        int col = idx % 136;
        int t   = idx / 136;               // c*4 + rr
        int rr  = t & 3;
        int c   = t >> 2;
        int hin = h0 + rr - 1;
        int win = col - 1;
        float v = 0.f;
        if (hin >= 0 && hin < IMG_H && win >= 0 && win < IMG_W) {
            float xv = h1[(((size_t)b * 8 + c) * IMG_H + hin) * IMG_W + win];
            xv = xv * sc1[c] + sh1[c];
            v = (xv >= 0.f) ? xv : alpha * xv;
        }
        sAh[(rr * 136 + col) * 8 + c] = (_Float16)v;
    }
    // stage weights: [n][k = tap*8 + c], taps padded 9 -> 12 with zeros
    for (int idx = tid; idx < 16 * 96; idx += 256) {
        int n = idx / 96;
        int k = idx % 96;
        int tap = k >> 3, c = k & 7;
        float v = (tap < 9) ? w2[(n * 8 + c) * 9 + tap] : 0.f;
        sWh[idx] = (_Float16)v;
    }
    __syncthreads();

    int lane = tid & 31;
    int wave = tid >> 5;
    int hi = (lane >> 4) & 1;
    int r_off = (wave >> 2) * 136;         // which of the 2 output rows
    int m0 = (wave & 3) * 32 + (lane & 15);
    int nc = lane & 15;

    v8f acc0 = {0.f,0.f,0.f,0.f,0.f,0.f,0.f,0.f};
    v8f acc1 = {0.f,0.f,0.f,0.f,0.f,0.f,0.f,0.f};
    #pragma unroll
    for (int j = 0; j < 3; ++j) {
        int pA0 = (hi ? TAP2(4*j+1) : TAP2(4*j))   + r_off + m0;
        int pA1 = (hi ? TAP2(4*j+3) : TAP2(4*j+2)) + r_off + m0;
        h8 a00 = sA8[pA0];
        h8 a01 = sA8[pA1];
        h8 a10 = sA8[pA0 + 16];
        h8 a11 = sA8[pA1 + 16];
        int bi = nc * 12 + j * 4 + hi;
        h8 b0 = sW8[bi];
        h8 b1 = sW8[bi + 2];
        v16h bf = SHUF16(b0, b1);
        acc0 = __builtin_amdgcn_wmma_f32_16x16x32_f16(
            false, SHUF16(a00, a01), false, bf, (short)0, acc0, false, false);
        acc1 = __builtin_amdgcn_wmma_f32_16x16x32_f16(
            false, SHUF16(a10, a11), false, bf, (short)0, acc1, false, false);
    }

    int rowOut = h0 + (wave >> 2);
    float bias = b2[nc];
    float s1 = 0.f, s2 = 0.f;
    int mrow = hi * 8;
    float* orow = h2 + (((size_t)b * 16 + nc) * IMG_H + rowOut) * IMG_W + (wave & 3) * 32;
    #pragma unroll
    for (int i = 0; i < 8; ++i) {
        float v0 = acc0[i] + bias;
        float v1 = acc1[i] + bias;
        orow[mrow + i]      = v0;
        orow[16 + mrow + i] = v1;
        s1 += v0 + v1;
        s2 += v0 * v0 + v1 * v1;
    }
    atomicAdd(&sstat[nc], s1);
    atomicAdd(&sstat[16 + nc], s2);
    __syncthreads();
    if (tid < 32) atomicAdd(&acc2[tid], sstat[tid]);   // 16 sums then 16 sumsq
}

// ---------------------------------------------------------------------------
// conv3: 16->32, k=9, pad=4, groups=2 — WMMA f16, channel-interleaved LDS.
// One block = (b, row-pair, g): stages 10 rows, computes 2 output rows.
// K = 81 taps x 8 ch = 648 -> 21 chunks of 32; 2 interleaved acc chains/wave.
// ---------------------------------------------------------------------------
__global__ __launch_bounds__(256) void conv3_wmma_k(
    const float* __restrict__ h2, const float* __restrict__ w3,
    const float* __restrict__ b3, const float* __restrict__ sc2,
    const float* __restrict__ sh2, const float* __restrict__ a2,
    float* __restrict__ h3, float* __restrict__ acc3)
{
    __shared__ h8       sA8[10 * 136];     // [row][col] -> 8 channels (f16)
    __shared__ h8       sW8[16 * 84];      // [n][tap(pad 84)] -> 8 channels
    __shared__ float    sstat[32];

    int blk = blockIdx.x;                  // ((b*64)+hp)*2 + g
    int g  = blk & 1;
    int hp = (blk >> 1) & 63;
    int b  = blk >> 7;
    int h0 = hp * 2;
    int tid = threadIdx.x;
    float alpha = a2[0];

    if (tid < 32) sstat[tid] = 0.f;

    _Float16* sAh = (_Float16*)sA8;
    _Float16* sWh = (_Float16*)sW8;

    // stage activations: 10 rows (h0-4 .. h0+5) x 136 cols x 8 ch, BN2+PReLU
    for (int idx = tid; idx < 10 * 136 * 8; idx += 256) {
        int col = idx % 136;
        int t   = idx / 136;               // c*10 + rr
        int rr  = t % 10;
        int c   = t / 10;
        int hin = h0 + rr - 4;
        int win = col - 4;
        float v = 0.f;
        if (hin >= 0 && hin < IMG_H && win >= 0 && win < IMG_W) {
            int cin = g * 8 + c;
            float xv = h2[(((size_t)b * 16 + cin) * IMG_H + hin) * IMG_W + win];
            xv = xv * sc2[cin] + sh2[cin];
            v = (xv >= 0.f) ? xv : alpha * xv;
        }
        sAh[(rr * 136 + col) * 8 + c] = (_Float16)v;
    }
    // stage weights: [n][k = tap*8 + c], taps padded 81 -> 84 with zeros
    for (int idx = tid; idx < 16 * 672; idx += 256) {
        int n = idx / 672;
        int k = idx % 672;
        int tap = k >> 3, c = k & 7;
        float v = (tap < 81) ? w3[((size_t)(g * 16 + n) * 8 + c) * 81 + tap] : 0.f;
        sWh[idx] = (_Float16)v;
    }
    __syncthreads();

    int lane = tid & 31;
    int wave = tid >> 5;
    int hi = (lane >> 4) & 1;
    int r_off = (wave >> 2) * 136;
    int m0 = (wave & 3) * 32 + (lane & 15);
    int nc = lane & 15;

    v8f acc0 = {0.f,0.f,0.f,0.f,0.f,0.f,0.f,0.f};
    v8f acc1 = {0.f,0.f,0.f,0.f,0.f,0.f,0.f,0.f};
    #pragma unroll
    for (int j = 0; j < 21; ++j) {
        int pA0 = (hi ? TAP3(4*j+1) : TAP3(4*j))   + r_off + m0;
        int pA1 = (hi ? TAP3(4*j+3) : TAP3(4*j+2)) + r_off + m0;
        h8 a00 = sA8[pA0];
        h8 a01 = sA8[pA1];
        h8 a10 = sA8[pA0 + 16];
        h8 a11 = sA8[pA1 + 16];
        int bi = nc * 84 + j * 4 + hi;
        h8 b0 = sW8[bi];
        h8 b1 = sW8[bi + 2];
        v16h bf = SHUF16(b0, b1);
        acc0 = __builtin_amdgcn_wmma_f32_16x16x32_f16(
            false, SHUF16(a00, a01), false, bf, (short)0, acc0, false, false);
        acc1 = __builtin_amdgcn_wmma_f32_16x16x32_f16(
            false, SHUF16(a10, a11), false, bf, (short)0, acc1, false, false);
    }

    int rowOut = h0 + (wave >> 2);
    int o = g * 16 + nc;
    float bias = b3[o];
    float s1 = 0.f, s2 = 0.f;
    int mrow = hi * 8;
    float* orow = h3 + (((size_t)b * 32 + o) * IMG_H + rowOut) * IMG_W + (wave & 3) * 32;
    #pragma unroll
    for (int i = 0; i < 8; ++i) {
        float v0 = acc0[i] + bias;
        float v1 = acc1[i] + bias;
        orow[mrow + i]      = v0;
        orow[16 + mrow + i] = v1;
        s1 += v0 + v1;
        s2 += v0 * v0 + v1 * v1;
    }
    atomicAdd(&sstat[nc], s1);
    atomicAdd(&sstat[16 + nc], s2);
    __syncthreads();
    if (tid < 16) {
        atomicAdd(&acc3[g * 16 + tid], sstat[tid]);
    } else if (tid < 32) {
        atomicAdd(&acc3[32 + g * 16 + (tid - 16)], sstat[tid]);
    }
}

// ---------------------------------------------------------------------------
// final: BN3+PReLU, locally-connected 32->2, ELU + (1+1e-7)
// ---------------------------------------------------------------------------
__global__ __launch_bounds__(256) void final_lc_k(
    const float* __restrict__ h3, const float* __restrict__ lc_w,
    const float* __restrict__ sc3, const float* __restrict__ sh3,
    const float* __restrict__ a3, float* __restrict__ out)
{
    int idx = blockIdx.x * 256 + threadIdx.x;     // over B*HW
    int pix = idx & (HW - 1);
    int b   = idx >> 14;
    float alpha = a3[0];
    float o0 = 0.f, o1 = 0.f;
    #pragma unroll 4
    for (int c = 0; c < 32; ++c) {
        float v = h3[((size_t)b * 32 + c) * HW + pix];
        v = v * sc3[c] + sh3[c];
        v = (v >= 0.f) ? v : alpha * v;
        o0 += v * lc_w[c * HW + pix];
        o1 += v * lc_w[(32 + c) * HW + pix];
    }
    float e0 = (o0 > 0.f) ? o0 : (expf(o0) - 1.f);
    float e1 = (o1 > 0.f) ? o1 : (expf(o1) - 1.f);
    out[((size_t)b * 2 + 0) * HW + pix] = e0 + ELU_ADD;
    out[((size_t)b * 2 + 1) * HW + pix] = e1 + ELU_ADD;
}

// ---------------------------------------------------------------------------
extern "C" void kernel_launch(void* const* d_in, const int* in_sizes, int n_in,
                              void* d_out, int out_size, void* d_ws, size_t ws_size,
                              hipStream_t stream) {
    const float* x    = (const float*)d_in[0];
    const float* w1   = (const float*)d_in[1];
    const float* b1   = (const float*)d_in[2];
    const float* g1   = (const float*)d_in[3];
    const float* be1  = (const float*)d_in[4];
    const float* a1   = (const float*)d_in[5];
    const float* w2   = (const float*)d_in[6];
    const float* b2   = (const float*)d_in[7];
    const float* g2   = (const float*)d_in[8];
    const float* be2  = (const float*)d_in[9];
    const float* a2   = (const float*)d_in[10];
    const float* w3   = (const float*)d_in[11];
    const float* b3   = (const float*)d_in[12];
    const float* g3   = (const float*)d_in[13];
    const float* be3  = (const float*)d_in[14];
    const float* a3   = (const float*)d_in[15];
    const float* lc_w = (const float*)d_in[16];
    float* out = (float*)d_out;

    float* ws   = (float*)d_ws;
    float* acc1 = ws;          // 16
    float* acc2 = ws + 16;     // 32
    float* acc3 = ws + 48;     // 64
    float* sc1  = ws + 112;    // 8
    float* sh1  = ws + 120;    // 8
    float* sc2  = ws + 128;    // 16
    float* sh2  = ws + 144;    // 16
    float* sc3  = ws + 160;    // 32
    float* sh3  = ws + 192;    // 32
    float* h1   = ws + 256;                          // B*8*HW
    float* h2   = h1 + (size_t)BATCH * 8 * HW;       // B*16*HW
    float* h3   = h2 + (size_t)BATCH * 16 * HW;      // B*32*HW

    init_acc_k<<<1, 128, 0, stream>>>(ws);

    conv1_k<<<dim3(BATCH * 8, HW / 256), 256, 0, stream>>>(x, w1, b1, h1, acc1);
    finalize_k<<<1, 32, 0, stream>>>(acc1, g1, be1, sc1, sh1, 8);

    conv2_wmma_k<<<BATCH * (IMG_H / 2), 256, 0, stream>>>(h1, w2, b2, sc1, sh1, a1, h2, acc2);
    finalize_k<<<1, 32, 0, stream>>>(acc2, g2, be2, sc2, sh2, 16);

    conv3_wmma_k<<<BATCH * (IMG_H / 2) * 2, 256, 0, stream>>>(h2, w3, b3, sc2, sh2, a2, h3, acc3);
    finalize_k<<<1, 32, 0, stream>>>(acc3, g3, be3, sc3, sh3, 32);

    final_lc_k<<<(BATCH * HW) / 256, 256, 0, stream>>>(h3, lc_w, sc3, sh3, a3, out);
}
